// PositionwiseFeedForward_76106820485308
// MI455X (gfx1250) — matrix operand order repaired and checked by reading, not compiled, not run on hardware
//
#include <hip/hip_runtime.h>
#include <hip/hip_bf16.h>

// ---------------- problem constants (from reference) ----------------
#define E_    24      // experts
#define D_    1024    // d_in
#define F_    4096    // d_hid
#define H_    4       // groups
#define B_    4       // batch
#define SEQ_  2048    // sequence length
#define NPG   512     // tokens per (h, b) = SEQ_/H_
#define S_    2048    // tokens per group = B_*NPG
#define CAP   171     // top-2 capacity = ceil(2*S/E)
#define CPAD  176     // capacity padded to multiple of 16 (11 m-tiles)
#define MT    11      // CPAD/16
#define KCHUNK 1024   // A-tile K chunk staged in LDS
#define ASTRIDE 1032  // LDS row stride (bf16) = 516 dwords -> uniform 2-way bank pattern

typedef __attribute__((ext_vector_type(16))) __bf16 bf16x16;
typedef __attribute__((ext_vector_type(8)))  __bf16 bf16x8;
typedef __attribute__((ext_vector_type(8)))  float  f32x8;

union FragAB { bf16x16 v; bf16x8 h[2]; };

static __device__ inline __bf16 f2bf(float f) { return (__bf16)f; }

#if defined(__has_builtin)
#if __has_builtin(__builtin_amdgcn_global_load_async_to_lds_b128) && \
    __has_builtin(__builtin_amdgcn_s_wait_asynccnt)
#define HAVE_ASYNC_LDS 1
#endif
#endif

#ifdef HAVE_ASYNC_LDS
// builtin expects: (int4 in AS(1)*, int4 in AS(3)*, imm offset, imm cpol)
typedef int v4i_ __attribute__((__vector_size__(16)));
typedef v4i_ __attribute__((address_space(1)))* gas_ptr;
typedef v4i_ __attribute__((address_space(3)))* las_ptr;
#endif

// ---------------------------------------------------------------
// Transpose fp32 [E][K][N] -> bf16 [E][N][K]  (N-major for WMMA B frags)
// ---------------------------------------------------------------
__global__ void transpose_to_bf16(const float* __restrict__ src,
                                  __bf16* __restrict__ dst, int K, int N) {
  __shared__ float tile[32][33];
  const int e  = blockIdx.z;
  const int nb = blockIdx.x * 32;
  const int kb = blockIdx.y * 32;
  const float* s = src + (size_t)e * K * N;
  __bf16*      d = dst + (size_t)e * K * N;
  const int tx = threadIdx.x, ty = threadIdx.y;   // block (32, 8)
#pragma unroll
  for (int j = 0; j < 32; j += 8)
    tile[ty + j][tx] = s[(size_t)(kb + ty + j) * N + (nb + tx)];
  __syncthreads();
#pragma unroll
  for (int j = 0; j < 32; j += 8)
    d[(size_t)(nb + ty + j) * K + (kb + tx)] = f2bf(tile[tx][ty + j]);
}

// ---------------------------------------------------------------
// Gating: logits = x @ Wg, softmax, top-2 (argmax + masked argmax)
// ---------------------------------------------------------------
__global__ void gate_kernel(const float* __restrict__ x,
                            const float* __restrict__ Wg,
                            int* __restrict__ idx1, int* __restrict__ idx2,
                            float* __restrict__ g1o, float* __restrict__ g2o) {
  const int gid = blockIdx.x;           // h*S + s
  const int h = gid / S_, s = gid % S_;
  const int b = s / NPG, i = s % NPG;
  const float* xt = x + ((size_t)b * SEQ_ + (size_t)h * NPG + i) * D_;
  const int tid = threadIdx.x;

  float acc[E_];
#pragma unroll
  for (int e = 0; e < E_; ++e) acc[e] = 0.f;
  for (int d = tid; d < D_; d += 256) {
    const float xv = xt[d];
    const float* wr = Wg + (size_t)d * E_;
#pragma unroll
    for (int e = 0; e < E_; ++e) acc[e] += xv * wr[e];
  }
  __shared__ float part[E_][257];
#pragma unroll
  for (int e = 0; e < E_; ++e) part[e][tid] = acc[e];
  __syncthreads();
  if (tid < E_) {
    float sum = 0.f;
    for (int t = 0; t < 256; ++t) sum += part[tid][t];
    part[tid][256] = sum;
  }
  __syncthreads();
  if (tid == 0) {
    float lg[E_];
#pragma unroll
    for (int e = 0; e < E_; ++e) lg[e] = part[e][256];
    int i1 = 0; float mx = lg[0];
#pragma unroll
    for (int e = 1; e < E_; ++e) if (lg[e] > mx) { mx = lg[e]; i1 = e; }
    float se = 0.f, ge[E_];
#pragma unroll
    for (int e = 0; e < E_; ++e) { ge[e] = __expf(lg[e] - mx); se += ge[e]; }
    int i2 = (i1 == 0) ? 1 : 0; float mx2 = lg[i2];
#pragma unroll
    for (int e = 0; e < E_; ++e)
      if (e != i1 && lg[e] > mx2) { mx2 = lg[e]; i2 = e; }
    idx1[gid] = i1; idx2[gid] = i2;
    g1o[gid] = ge[i1] / se; g2o[gid] = ge[i2] / se;
  }
}

// ---------------------------------------------------------------
// Sequential capacity scan: 1 wave per group; lane = expert.
// ---------------------------------------------------------------
__global__ void assign_kernel(const int* __restrict__ idx1,
                              const int* __restrict__ idx2,
                              const float* __restrict__ g1i,
                              const float* __restrict__ g2i,
                              int* __restrict__ p1o, int* __restrict__ p2o,
                              float* __restrict__ g1o, float* __restrict__ g2o) {
  const int h = blockIdx.x;
  const int lane = threadIdx.x;          // blockDim = 32
  __shared__ int p1s[S_];
  __shared__ int p2s[S_];
  const int* i1 = idx1 + (size_t)h * S_;
  const int* i2 = idx2 + (size_t)h * S_;
  int cnt = 0;
  for (int s = 0; s < S_; ++s) {
    if (lane == i1[s]) { p1s[s] = cnt; ++cnt; }
  }
  for (int s = 0; s < S_; ++s) {
    if (lane == i2[s]) { p2s[s] = cnt; ++cnt; }
  }
  __syncthreads();
  for (int s = lane; s < S_; s += 32) {
    const size_t g = (size_t)h * S_ + s;
    int q1 = p1s[s], q2 = p2s[s];
    float g1 = g1i[g], g2 = g2i[g];
    if (q1 >= CAP) g1 = 0.f;
    if (q2 >= CAP) g2 = 0.f;
    const float den = fmaxf(g1 + g2, 1e-9f);
    g1o[g] = g1 / den; g2o[g] = g2 / den;
    p1o[g] = q1;       p2o[g] = q2;
  }
}

// ---------------------------------------------------------------
// Scatter tokens (fp32 -> bf16) into expert buffers [H][E][CPAD][D]
// ---------------------------------------------------------------
__global__ void scatter_x(const float* __restrict__ x,
                          const int* __restrict__ idx1, const int* __restrict__ idx2,
                          const int* __restrict__ p1, const int* __restrict__ p2,
                          __bf16* __restrict__ Xe) {
  const int gid = blockIdx.x;
  const int h = gid / S_, s = gid % S_;
  const int b = s / NPG, i = s % NPG;
  const float* xt = x + ((size_t)b * SEQ_ + (size_t)h * NPG + i) * D_;
  const int tp = threadIdx.x;            // 256 threads, 4 elems each
  const float4 v = ((const float4*)xt)[tp];
  const __bf16 b0 = f2bf(v.x), b1 = f2bf(v.y), b2 = f2bf(v.z), b3 = f2bf(v.w);
  const int e1 = idx1[gid], e2 = idx2[gid];
  const int q1 = p1[gid],  q2 = p2[gid];
  if (q1 < CAP) {
    __bf16* d = Xe + (((size_t)(h * E_ + e1)) * CPAD + q1) * D_ + tp * 4;
    d[0] = b0; d[1] = b1; d[2] = b2; d[3] = b3;
  }
  if (q2 < CAP) {
    __bf16* d = Xe + (((size_t)(h * E_ + e2)) * CPAD + q2) * D_ + tp * 4;
    d[0] = b0; d[1] = b1; d[2] = b2; d[3] = b3;
  }
}

// ---------------------------------------------------------------
// Batched expert GEMM via v_wmma_f32_16x16x32_bf16.
//   A  : bf16 [Z][CPAD][K]   (Z = h*E + e)
//   BT : bf16 [E][N][K]      (N-major so B frags are contiguous)
//   Dq : fp32 [Z][CPAD][N]   (+ bias[e][N])
// grid = (MT, N/512, H*E); block = 256 (8 waves); wave covers 4 n-tiles.
// A tile (16 x KCHUNK) staged in LDS via async-to-LDS, shared by 8 waves.
// ---------------------------------------------------------------
__global__ void moe_gemm(const __bf16* __restrict__ A,
                         const __bf16* __restrict__ BT,
                         const float* __restrict__ bias,
                         float* __restrict__ Dq, int K, int N) {
  __shared__ __attribute__((aligned(16))) __bf16 As[16 * ASTRIDE];

  const int z    = blockIdx.z;
  const int e    = z % E_;
  const int m0   = blockIdx.x * 16;
  const int wave = threadIdx.x >> 5;
  const int lane = threadIdx.x & 31;
  const int half = lane >> 4;
  const int r    = lane & 15;
  const int nbase = blockIdx.y * 512 + wave * 64;
  const int tid  = threadIdx.x;

  const __bf16* Abase = A  + ((size_t)z * CPAD + m0) * K;
  const __bf16* Bbase = BT + (size_t)e * (size_t)N * K;

  f32x8 acc[4];
#pragma unroll
  for (int j = 0; j < 4; ++j)
#pragma unroll
    for (int i = 0; i < 8; ++i) acc[j][i] = 0.f;

  for (int kc = 0; kc < K; kc += KCHUNK) {
    // ---- stage A chunk [16][KCHUNK] into LDS (2048 x 16B transfers) ----
#pragma unroll
    for (int it = 0; it < (16 * KCHUNK / 8) / 256; ++it) {
      const int l   = tid + it * 256;       // 16B-chunk index
      const int row = l >> 7;               // KCHUNK/8 = 128 chunks per row
      const int ck  = l & 127;
      const __bf16* src = Abase + (size_t)row * K + kc + ck * 8;
      __bf16*       dst = &As[row * ASTRIDE + ck * 8];
#ifdef HAVE_ASYNC_LDS
      __builtin_amdgcn_global_load_async_to_lds_b128(
          (gas_ptr)(uintptr_t)src, (las_ptr)(unsigned)(uintptr_t)dst, 0, 0);
#else
      *(bf16x8*)dst = *(const bf16x8*)src;
#endif
    }
#ifdef HAVE_ASYNC_LDS
    __builtin_amdgcn_s_wait_asynccnt(0);
#endif
    __syncthreads();

    // ---- consume the chunk ----
    for (int k0 = 0; k0 < KCHUNK; k0 += 32) {
      // A frag from LDS: 16-bit A 16x32 layout (row = r, chunks at half*8, 16+half*8)
      const __bf16* Ap = &As[r * ASTRIDE + k0 + half * 8];
      FragAB a;
      a.h[0] = *(const bf16x8*)(Ap);
      a.h[1] = *(const bf16x8*)(Ap + 16);
      // B frags (all 4 n-tiles) buffered in registers before the WMMAs
      FragAB bfr[4];
#pragma unroll
      for (int j = 0; j < 4; ++j) {
        const __bf16* Bp = Bbase + (size_t)(nbase + j * 16 + r) * K
                         + kc + k0 + half * 8;
        bfr[j].h[0] = *(const bf16x8*)(Bp);
        bfr[j].h[1] = *(const bf16x8*)(Bp + 16);
        __builtin_prefetch(Bp + KCHUNK, 0, 1);   // speculative: next chunk's row
      }
#pragma unroll
      for (int j = 0; j < 4; ++j)
        acc[j] = __builtin_amdgcn_wmma_f32_16x16x32_bf16(
            false, a.v, false, bfr[j].v, (short)0, acc[j], false, false);
    }
    __syncthreads();
  }

  float* Dp = Dq + (size_t)z * CPAD * N;
#pragma unroll
  for (int j = 0; j < 4; ++j) {
    const int nn = nbase + j * 16 + r;
    const float bi = bias ? bias[(size_t)e * N + nn] : 0.f;
#pragma unroll
    for (int i = 0; i < 8; ++i)   // VGPR i: M = i (lanes 0-15) / 8+i (lanes 16-31)
      Dp[(size_t)(m0 + half * 8 + i) * N + nn] = acc[j][i] + bi;
  }
}

// ---------------------------------------------------------------
// Combine expert GEMM1 outputs at token level, LayerNorm(d_hid), ReLU,
// re-scatter (bf16) to stage-2 expert buffers [H][E][CPAD][F]
// ---------------------------------------------------------------
__global__ void combine1_kernel(const float* __restrict__ H1,
                                const int* __restrict__ idx1, const int* __restrict__ idx2,
                                const int* __restrict__ p1, const int* __restrict__ p2,
                                const float* __restrict__ g1, const float* __restrict__ g2,
                                const float* __restrict__ lnw, const float* __restrict__ lnb,
                                __bf16* __restrict__ Xe2) {
  const int gid = blockIdx.x;
  const int h = gid / S_;
  const int tid = threadIdx.x;
  const int e1 = idx1[gid], e2 = idx2[gid];
  const int q1 = p1[gid],  q2 = p2[gid];
  const float w1 = g1[gid], w2 = g2[gid];
  const float* ha = (q1 < CAP) ? H1 + (((size_t)(h * E_ + e1)) * CPAD + q1) * F_ : nullptr;
  const float* hb = (q2 < CAP) ? H1 + (((size_t)(h * E_ + e2)) * CPAD + q2) * F_ : nullptr;

  float v[16];
  float sum = 0.f, sq = 0.f;
#pragma unroll
  for (int j = 0; j < 16; ++j) {
    const int f = tid + j * 256;
    float t = 0.f;
    if (ha) t += w1 * ha[f];
    if (hb) t += w2 * hb[f];
    v[j] = t; sum += t; sq += t * t;
  }
  __shared__ float rs[256], rq[256];
  rs[tid] = sum; rq[tid] = sq;
  __syncthreads();
  for (int st = 128; st > 0; st >>= 1) {
    if (tid < st) { rs[tid] += rs[tid + st]; rq[tid] += rq[tid + st]; }
    __syncthreads();
  }
  const float mean = rs[0] * (1.0f / F_);
  const float var  = rq[0] * (1.0f / F_) - mean * mean;
  const float rstd = rsqrtf(var + 1e-6f);
  __bf16* d1 = (q1 < CAP) ? Xe2 + (((size_t)(h * E_ + e1)) * CPAD + q1) * F_ : nullptr;
  __bf16* d2 = (q2 < CAP) ? Xe2 + (((size_t)(h * E_ + e2)) * CPAD + q2) * F_ : nullptr;
  const float* lw = lnw + (size_t)h * F_;
  const float* lb = lnb + (size_t)h * F_;
#pragma unroll
  for (int j = 0; j < 16; ++j) {
    const int f = tid + j * 256;
    float y = (v[j] - mean) * rstd * lw[f] + lb[f];
    y = fmaxf(y, 0.f);
    const __bf16 bo = f2bf(y);
    if (d1) d1[f] = bo;
    if (d2) d2[f] = bo;
  }
}

// ---------------------------------------------------------------
// Combine GEMM2 outputs, add residual x, final LayerNorm -> d_out
// ---------------------------------------------------------------
__global__ void final_kernel(const float* __restrict__ H2,
                             const float* __restrict__ x,
                             const int* __restrict__ idx1, const int* __restrict__ idx2,
                             const int* __restrict__ p1, const int* __restrict__ p2,
                             const float* __restrict__ g1, const float* __restrict__ g2,
                             const float* __restrict__ lnw, const float* __restrict__ lnb,
                             float* __restrict__ out) {
  const int gid = blockIdx.x;
  const int h = gid / S_, s = gid % S_;
  const int b = s / NPG, i = s % NPG;
  const size_t tok = ((size_t)b * SEQ_ + (size_t)h * NPG + i) * D_;
  const int tid = threadIdx.x;
  const int e1 = idx1[gid], e2 = idx2[gid];
  const int q1 = p1[gid],  q2 = p2[gid];
  const float w1 = g1[gid], w2 = g2[gid];
  const float* ha = (q1 < CAP) ? H2 + (((size_t)(h * E_ + e1)) * CPAD + q1) * D_ : nullptr;
  const float* hb = (q2 < CAP) ? H2 + (((size_t)(h * E_ + e2)) * CPAD + q2) * D_ : nullptr;

  float v[4];
  float sum = 0.f, sq = 0.f;
#pragma unroll
  for (int j = 0; j < 4; ++j) {
    const int d = tid + j * 256;
    float t = x[tok + d];
    if (ha) t += w1 * ha[d];
    if (hb) t += w2 * hb[d];
    v[j] = t; sum += t; sq += t * t;
  }
  __shared__ float rs[256], rq[256];
  rs[tid] = sum; rq[tid] = sq;
  __syncthreads();
  for (int st = 128; st > 0; st >>= 1) {
    if (tid < st) { rs[tid] += rs[tid + st]; rq[tid] += rq[tid + st]; }
    __syncthreads();
  }
  const float mean = rs[0] * (1.0f / D_);
  const float var  = rq[0] * (1.0f / D_) - mean * mean;
  const float rstd = rsqrtf(var + 1e-6f);
#pragma unroll
  for (int j = 0; j < 4; ++j) {
    const int d = tid + j * 256;
    out[tok + d] = (v[j] - mean) * rstd * lnw[d] + lnb[d];
  }
}

// ---------------------------------------------------------------
extern "C" void kernel_launch(void* const* d_in, const int* in_sizes, int n_in,
                              void* d_out, int out_size, void* d_ws, size_t ws_size,
                              hipStream_t stream) {
  (void)in_sizes; (void)n_in; (void)out_size; (void)ws_size;
  const float* x   = (const float*)d_in[0];
  const float* Wg  = (const float*)d_in[1];
  const float* W1  = (const float*)d_in[2];
  const float* b1  = (const float*)d_in[3];
  const float* W2  = (const float*)d_in[4];
  const float* b2  = (const float*)d_in[5];
  const float* lhw = (const float*)d_in[6];
  const float* lhb = (const float*)d_in[7];
  const float* low = (const float*)d_in[8];
  const float* lob = (const float*)d_in[9];
  float* out = (float*)d_out;

  char* ws = (char*)d_ws;
  size_t off = 0;
  auto alloc = [&](size_t bytes) -> void* {
    void* p = ws + off;
    off += (bytes + 255) & ~(size_t)255;
    return p;
  };

  __bf16* w1t = (__bf16*)alloc((size_t)E_ * F_ * D_ * 2);         // [E][F][D]
  __bf16* w2t = (__bf16*)alloc((size_t)E_ * D_ * F_ * 2);         // [E][D][F]
  __bf16* Xe  = (__bf16*)alloc((size_t)H_ * E_ * CPAD * D_ * 2);  // stage1 A
  float*  H1  = (float*) alloc((size_t)H_ * E_ * CPAD * F_ * 4);  // stage1 out
  __bf16* Xe2 = (__bf16*)alloc((size_t)H_ * E_ * CPAD * F_ * 2);  // stage2 A
  float*  H2  = (float*) alloc((size_t)H_ * E_ * CPAD * D_ * 4);  // stage2 out
  int*   idx1 = (int*)  alloc((size_t)H_ * S_ * 4);
  int*   idx2 = (int*)  alloc((size_t)H_ * S_ * 4);
  int*   p1   = (int*)  alloc((size_t)H_ * S_ * 4);
  int*   p2   = (int*)  alloc((size_t)H_ * S_ * 4);
  float* g1r  = (float*)alloc((size_t)H_ * S_ * 4);
  float* g2r  = (float*)alloc((size_t)H_ * S_ * 4);
  float* g1n  = (float*)alloc((size_t)H_ * S_ * 4);
  float* g2n  = (float*)alloc((size_t)H_ * S_ * 4);

  // weights -> bf16, N-major
  transpose_to_bf16<<<dim3(F_ / 32, D_ / 32, E_), dim3(32, 8), 0, stream>>>(W1, w1t, D_, F_);
  transpose_to_bf16<<<dim3(D_ / 32, F_ / 32, E_), dim3(32, 8), 0, stream>>>(W2, w2t, F_, D_);

  // gating + capacity scan
  gate_kernel<<<H_ * S_, 256, 0, stream>>>(x, Wg, idx1, idx2, g1r, g2r);
  assign_kernel<<<H_, 32, 0, stream>>>(idx1, idx2, g1r, g2r, p1, p2, g1n, g2n);

  // dispatch -> expert GEMM1 -> combine/LN/ReLU -> re-dispatch -> expert GEMM2
  scatter_x<<<H_ * S_, 256, 0, stream>>>(x, idx1, idx2, p1, p2, Xe);
  moe_gemm<<<dim3(MT, F_ / 512, H_ * E_), 256, 0, stream>>>(Xe, w1t, b1, H1, D_, F_);
  combine1_kernel<<<H_ * S_, 256, 0, stream>>>(H1, idx1, idx2, p1, p2, g1n, g2n,
                                               lhw, lhb, Xe2);
  moe_gemm<<<dim3(MT, D_ / 512, H_ * E_), 256, 0, stream>>>(Xe2, w2t, b2, H2, F_, D_);

  // combine + residual + final LN
  final_kernel<<<H_ * S_, 256, 0, stream>>>(H2, x, idx1, idx2, p1, p2, g1n, g2n,
                                            low, lob, out);
}
